// MaskDINOEncoder_6588479832640
// MI455X (gfx1250) — matrix operand order repaired and checked by reading, not compiled, not run on hardware
//
#include <hip/hip_runtime.h>
#include <hip/hip_bf16.h>

// ---------------------------------------------------------------------------
// MaskDINO deformable encoder, MI455X (gfx1250, wave32, WMMA).
// GEMMs run on v_wmma_f32_16x16x32_bf16 (fp32 accum). LDS tiles that need no
// dtype conversion are staged with GLOBAL_LOAD_ASYNC_TO_LDS (ASYNCcnt) when
// the toolchain exposes the builtin. Everything else is bandwidth-trivial
// VALU work that lives in the 192MB L2.
// ---------------------------------------------------------------------------

#define LQ_TOT 21760
#define DMODEL 256
#define NHEAD  8
#define NLVL   4
#define NPTS   4
#define DHEAD  32
#define NLAYER 6
#define DFFN   1024

typedef __attribute__((ext_vector_type(16))) __bf16 v16bf;
typedef __attribute__((ext_vector_type(8)))  __bf16 v8bf;
typedef __attribute__((ext_vector_type(8)))  float  v8f;
typedef __attribute__((ext_vector_type(4)))  int    v4i;

#define GLOBAL_AS __attribute__((address_space(1)))
#define SHARED_AS __attribute__((address_space(3)))

#if __has_builtin(__builtin_amdgcn_global_load_async_to_lds_b128) && \
    __has_builtin(__builtin_amdgcn_s_wait_asynccnt)
#define USE_ASYNC_LDS 1
#else
#define USE_ASYNC_LDS 0
#endif

// fp32 -> bf16, round-to-nearest-even
__device__ inline unsigned short f2bf1(float x) {
  unsigned int u = __float_as_uint(x);
  u = (u + 0x7FFFu + ((u >> 16) & 1u)) >> 16;
  return (unsigned short)u;
}
__device__ inline unsigned int f2bf_pack(float a, float b) {
  return (unsigned int)f2bf1(a) | ((unsigned int)f2bf1(b) << 16);
}

// Stage 16 contiguous bf16 (32B) global -> LDS. Async DMA when available.
__device__ inline void stage32_bf16(const unsigned short* __restrict__ src,
                                    unsigned short* dst) {
#if USE_ASYNC_LDS
  unsigned short* snc = const_cast<unsigned short*>(src);
  __builtin_amdgcn_global_load_async_to_lds_b128(
      (GLOBAL_AS v4i*)snc, (SHARED_AS v4i*)dst, 0, 0);
  __builtin_amdgcn_global_load_async_to_lds_b128(
      (GLOBAL_AS v4i*)snc, (SHARED_AS v4i*)dst, 16, 0);
#else
  uint4 u0 = ((const uint4*)src)[0];
  uint4 u1 = ((const uint4*)src)[1];
  *(uint4*)dst       = u0;
  *(uint4*)(dst + 8) = u1;
#endif
}

// ---------------------------------------------------------------------------
// Tiled WMMA GEMM:  C[M,N] = act( (A (+A2)) @ W^T + bias )
//   A  : f32 [M,K]  (or bf16 [M,K] when ABF16)
//   W  : bf16 [N,K] row-major (pre-converted)
//   block tile 128x128, BK=32, 256 threads = 8 waves (2x4), wave tile 64x32
//   N, K compile-time so the epilogue/store addressing folds to immediates.
// ---------------------------------------------------------------------------
template <int N, int K, bool ABF16, bool HASA2, bool RELU, bool OUTBF16>
__global__ __launch_bounds__(256) void gemm_kernel(
    const void* __restrict__ Ap, const float* __restrict__ A2,
    const unsigned short* __restrict__ Wb, const float* __restrict__ bias,
    void* __restrict__ Cp) {
  constexpr int LDA = 40;  // 32 + 8 pad (ushorts) to spread LDS banks
  constexpr int LDB = 40;
  __shared__ unsigned short As[128 * LDA];
  __shared__ unsigned short Bs[128 * LDB];

  const float*          Af = (const float*)Ap;
  const unsigned short* Ab = (const unsigned short*)Ap;
  float*                Cf = (float*)Cp;
  unsigned short*       Cb = (unsigned short*)Cp;

  const int tid   = threadIdx.x;
  const int wid   = tid >> 5;
  const int lane  = tid & 31;
  const int waveM = wid >> 2;   // 0..1  (64 rows each)
  const int waveN = wid & 3;    // 0..3  (32 cols each)
  const int lr    = lane & 15;
  const int half  = lane >> 4;

  const int mBase = blockIdx.y * 128;
  const int nBase = blockIdx.x * 128;

  // cooperative-load coordinates: 2 threads per row, 16 elems each
  const int lrow = tid >> 1;
  const int lcol = (tid & 1) * 16;

  const v8f vzero = {0.f, 0.f, 0.f, 0.f, 0.f, 0.f, 0.f, 0.f};
  v8f acc[4][2];
#pragma unroll
  for (int m = 0; m < 4; ++m)
#pragma unroll
    for (int n = 0; n < 2; ++n) acc[m][n] = vzero;

  union Frag { v16bf v; v8bf h[2]; };

  for (int kk = 0; kk < K; kk += 32) {
    // Prefetch next K-slab (lowers to global_prefetch on gfx1250)
    if (kk + 32 < K) {
      if (ABF16)
        __builtin_prefetch(Ab + (size_t)(mBase + lrow) * K + kk + 32 + lcol, 0, 3);
      else
        __builtin_prefetch(Af + (size_t)(mBase + lrow) * K + kk + 32 + lcol, 0, 3);
      __builtin_prefetch(Wb + (size_t)(nBase + lrow) * K + kk + 32 + lcol, 0, 3);
    }

    // ---- stage A tile (128x32 bf16) into LDS ----
    if constexpr (!ABF16) {
      // fp32 (+optional fused add) -> bf16 through VALU
      const float* ap = Af + (size_t)(mBase + lrow) * K + kk + lcol;
      float4 f[4];
#pragma unroll
      for (int i = 0; i < 4; ++i) f[i] = ((const float4*)ap)[i];
      if constexpr (HASA2) {
        const float* a2p = A2 + (size_t)(mBase + lrow) * K + kk + lcol;
#pragma unroll
        for (int i = 0; i < 4; ++i) {
          float4 g4 = ((const float4*)a2p)[i];
          f[i].x += g4.x; f[i].y += g4.y; f[i].z += g4.z; f[i].w += g4.w;
        }
      }
      unsigned int* dst = (unsigned int*)&As[lrow * LDA + lcol];
#pragma unroll
      for (int i = 0; i < 4; ++i) {
        dst[2 * i]     = f2bf_pack(f[i].x, f[i].y);
        dst[2 * i + 1] = f2bf_pack(f[i].z, f[i].w);
      }
    } else {
      // already bf16: async DMA straight into LDS
      stage32_bf16(Ab + (size_t)(mBase + lrow) * K + kk + lcol,
                   &As[lrow * LDA + lcol]);
    }
    // ---- stage W tile (128 out-channels x 32 bf16) into LDS ----
    stage32_bf16(Wb + (size_t)(nBase + lrow) * K + kk + lcol,
                 &Bs[lrow * LDB + lcol]);

#if USE_ASYNC_LDS
    __builtin_amdgcn_s_wait_asynccnt(0);
#endif
    __syncthreads();

    // ---- fragments per ISA 16-bit A (16x32) / B lane layouts ----
    Frag a[4], b[2];
#pragma unroll
    for (int m = 0; m < 4; ++m) {
      int row = waveM * 64 + m * 16 + lr;
      a[m].h[0] = *(const v8bf*)&As[row * LDA + half * 8];
      a[m].h[1] = *(const v8bf*)&As[row * LDA + 16 + half * 8];
    }
#pragma unroll
    for (int n = 0; n < 2; ++n) {
      int col = waveN * 32 + n * 16 + lr;
      b[n].h[0] = *(const v8bf*)&Bs[col * LDB + half * 8];
      b[n].h[1] = *(const v8bf*)&Bs[col * LDB + 16 + half * 8];
    }
#pragma unroll
    for (int m = 0; m < 4; ++m)
#pragma unroll
      for (int n = 0; n < 2; ++n)
        acc[m][n] = __builtin_amdgcn_wmma_f32_16x16x32_bf16(
            false, a[m].v, false, b[n].v, (short)0, acc[m][n], false, false);
    __syncthreads();
  }

  // ---- epilogue: C layout lanes0-15 -> M=e, lanes16-31 -> M=8+e ----
  const int cN   = lane & 15;
  const int rOff = half * 8;
#pragma unroll
  for (int n = 0; n < 2; ++n) {
    const int gcol = nBase + waveN * 32 + n * 16 + cN;
    const float bb = bias[gcol];
#pragma unroll
    for (int m = 0; m < 4; ++m) {
      const int row0 = mBase + waveM * 64 + m * 16 + rOff;
      if constexpr (OUTBF16) {
        unsigned short* cp = Cb + (size_t)row0 * N + gcol;
#pragma unroll
        for (int e = 0; e < 8; ++e) {
          float v = acc[m][n][e] + bb;
          if constexpr (RELU) v = fmaxf(v, 0.f);
          cp[(size_t)e * N] = f2bf1(v);  // e*N folds to an immediate offset
        }
      } else {
        float* cp = Cf + (size_t)row0 * N + gcol;
#pragma unroll
        for (int e = 0; e < 8; ++e) {
          float v = acc[m][n][e] + bb;
          if constexpr (RELU) v = fmaxf(v, 0.f);
          cp[(size_t)e * N] = v;
        }
      }
    }
  }
}

// ---------------------------------------------------------------------------
// NCHW -> [LQ,D] flattening, + level embed into pos, + reference points
// ---------------------------------------------------------------------------
__global__ void prep_kernel(const float* __restrict__ s0, const float* __restrict__ p0,
                            const float* __restrict__ s1, const float* __restrict__ p1,
                            const float* __restrict__ s2, const float* __restrict__ p2,
                            const float* __restrict__ s3, const float* __restrict__ p3,
                            const float* __restrict__ lev,
                            float* __restrict__ s, float* __restrict__ pos,
                            float* __restrict__ refp) {
  int gid = blockIdx.x * blockDim.x + threadIdx.x;
  if (gid >= LQ_TOT * DMODEL) return;
  int d = gid & (DMODEL - 1);
  int q = gid >> 8;
  int l, start, H, W;
  const float *sp, *pp;
  if (q < 16384)      { l = 0; start = 0;     H = 128; W = 128; sp = s0; pp = p0; }
  else if (q < 20480) { l = 1; start = 16384; H = 64;  W = 64;  sp = s1; pp = p1; }
  else if (q < 21504) { l = 2; start = 20480; H = 32;  W = 32;  sp = s2; pp = p2; }
  else                { l = 3; start = 21504; H = 16;  W = 16;  sp = s3; pp = p3; }
  int p  = q - start;
  int hw = H * W;
  s[(size_t)q * DMODEL + d]   = sp[(size_t)d * hw + p];
  pos[(size_t)q * DMODEL + d] = pp[(size_t)d * hw + p] + lev[l * DMODEL + d];
  if (d == 0) {
    int py = p / W, px = p % W;
    refp[q * 2 + 0] = ((float)px + 0.5f) / (float)W;
    refp[q * 2 + 1] = ((float)py + 0.5f) / (float)H;
  }
}

__global__ void cvt_bf16_kernel(const float* __restrict__ in,
                                unsigned short* __restrict__ out, int n) {
  int i = blockIdx.x * blockDim.x + threadIdx.x;
  if (i < n) out[i] = f2bf1(in[i]);
}

// softmax over the contiguous NL*NP=16 logits per (query, head); in place
__global__ void softmax16_kernel(float* __restrict__ aw) {
  int gid = blockIdx.x * blockDim.x + threadIdx.x;
  if (gid >= LQ_TOT * NHEAD) return;
  float* a = aw + (size_t)gid * 16;
  float mx = -1e30f;
#pragma unroll
  for (int i = 0; i < 16; ++i) mx = fmaxf(mx, a[i]);
  float e[16], s = 0.f;
#pragma unroll
  for (int i = 0; i < 16; ++i) { e[i] = __expf(a[i] - mx); s += e[i]; }
  float inv = 1.f / s;
#pragma unroll
  for (int i = 0; i < 16; ++i) a[i] = e[i] * inv;
}

// Multi-scale deformable attention sampling. One lane per head-channel
// (32 lanes == one head). Gather-bound; value tensor lives in L2.
__global__ void msda_kernel(const float* __restrict__ val, const float* __restrict__ off,
                            const float* __restrict__ aw, const float* __restrict__ refp,
                            float* __restrict__ out) {
  int gid = blockIdx.x * blockDim.x + threadIdx.x;
  if (gid >= LQ_TOT * NHEAD * DHEAD) return;
  int d  = gid & 31;
  int qh = gid >> 5;
  int h  = qh & 7;
  int q  = qh >> 3;
  float rx = refp[q * 2 + 0], ry = refp[q * 2 + 1];
  const float* offq = off + (size_t)q * DMODEL + h * (NLVL * NPTS * 2);
  const float* awq  = aw  + (size_t)q * (NHEAD * NLVL * NPTS) + h * (NLVL * NPTS);
  const int LS[4] = {0, 16384, 20480, 21504};
  const int LW[4] = {128, 64, 32, 16};
  float acc = 0.f;
#pragma unroll
  for (int l = 0; l < 4; ++l) {
    const int W = LW[l], H = LW[l], start = LS[l];
#pragma unroll
    for (int p = 0; p < 4; ++p) {
      float ox = offq[(l * 4 + p) * 2 + 0];
      float oy = offq[(l * 4 + p) * 2 + 1];
      float wA = awq[l * 4 + p];
      float xf = rx * (float)W + ox - 0.5f;
      float yf = ry * (float)H + oy - 0.5f;
      float x0f = floorf(xf), y0f = floorf(yf);
      int x0 = (int)x0f, y0 = (int)y0f;
      float fx = xf - x0f, fy = yf - y0f;
      float w00 = (1.f - fx) * (1.f - fy) * wA;
      float w10 = fx * (1.f - fy) * wA;
      float w01 = (1.f - fx) * fy * wA;
      float w11 = fx * fy * wA;
      const size_t hd = (size_t)h * DHEAD + d;
      if (x0 >= 0 && x0 < W && y0 >= 0 && y0 < H)
        acc += w00 * val[((size_t)(start + y0 * W + x0)) * DMODEL + hd];
      if (x0 + 1 >= 0 && x0 + 1 < W && y0 >= 0 && y0 < H)
        acc += w10 * val[((size_t)(start + y0 * W + x0 + 1)) * DMODEL + hd];
      if (x0 >= 0 && x0 < W && y0 + 1 >= 0 && y0 + 1 < H)
        acc += w01 * val[((size_t)(start + (y0 + 1) * W + x0)) * DMODEL + hd];
      if (x0 + 1 >= 0 && x0 + 1 < W && y0 + 1 >= 0 && y0 + 1 < H)
        acc += w11 * val[((size_t)(start + (y0 + 1) * W + x0 + 1)) * DMODEL + hd];
    }
  }
  out[(size_t)q * DMODEL + (size_t)h * DHEAD + d] = acc;
}

// s = LN(s + a) * g + b, one wave32 per row, shuffle reductions
__global__ void add_ln_kernel(float* __restrict__ s, const float* __restrict__ a,
                              const float* __restrict__ g, const float* __restrict__ b) {
  int gid  = blockIdx.x * blockDim.x + threadIdx.x;
  int lane = gid & 31;
  int row  = gid >> 5;
  if (row >= LQ_TOT) return;
  float x[8];
  float sum = 0.f, sq = 0.f;
#pragma unroll
  for (int i = 0; i < 8; ++i) {
    int idx = i * 32 + lane;
    float v = s[(size_t)row * DMODEL + idx] + a[(size_t)row * DMODEL + idx];
    x[i] = v; sum += v; sq += v * v;
  }
#pragma unroll
  for (int m = 16; m >= 1; m >>= 1) {
    sum += __shfl_xor(sum, m, 32);
    sq  += __shfl_xor(sq,  m, 32);
  }
  float mean = sum * (1.f / 256.f);
  float var  = sq * (1.f / 256.f) - mean * mean;
  float rstd = rsqrtf(var + 1e-5f);
#pragma unroll
  for (int i = 0; i < 8; ++i) {
    int idx = i * 32 + lane;
    s[(size_t)row * DMODEL + idx] = (x[i] - mean) * rstd * g[idx] + b[idx];
  }
}

// ---------------------------------------------------------------------------
extern "C" void kernel_launch(void* const* d_in, const int* in_sizes, int n_in,
                              void* d_out, int out_size, void* d_ws, size_t ws_size,
                              hipStream_t stream) {
  (void)in_sizes; (void)n_in; (void)out_size;
  const float* src0 = (const float*)d_in[0];
  const float* pos0 = (const float*)d_in[1];
  const float* src1 = (const float*)d_in[2];
  const float* pos1 = (const float*)d_in[3];
  const float* src2 = (const float*)d_in[4];
  const float* pos2 = (const float*)d_in[5];
  const float* src3 = (const float*)d_in[6];
  const float* pos3 = (const float*)d_in[7];
  const float* lev  = (const float*)d_in[8];
  const float* Woff = (const float*)d_in[9];
  const float* boff = (const float*)d_in[10];
  const float* Watt = (const float*)d_in[11];
  const float* batt = (const float*)d_in[12];
  const float* Wv   = (const float*)d_in[13];
  const float* bv   = (const float*)d_in[14];
  const float* Wo   = (const float*)d_in[15];
  const float* bo   = (const float*)d_in[16];
  const float* g1   = (const float*)d_in[17];
  const float* b1   = (const float*)d_in[18];
  const float* Wl1  = (const float*)d_in[19];
  const float* bl1  = (const float*)d_in[20];
  const float* Wl2  = (const float*)d_in[21];
  const float* bl2  = (const float*)d_in[22];
  const float* g2   = (const float*)d_in[23];
  const float* b2   = (const float*)d_in[24];

  char* ws = (char*)d_ws;
  size_t o = 0;
  auto take = [&](size_t bytes) -> char* {
    char* p = ws + o;
    o += (bytes + 255) & ~(size_t)255;
    return p;
  };
  float* sbuf  = (float*)take((size_t)LQ_TOT * DMODEL * 4);
  float* posem = (float*)take((size_t)LQ_TOT * DMODEL * 4);
  float* offb  = (float*)take((size_t)LQ_TOT * DMODEL * 4);
  float* attnb = (float*)take((size_t)LQ_TOT * 128 * 4);
  float* valb  = (float*)take((size_t)LQ_TOT * DMODEL * 4);
  float* msdab = (float*)take((size_t)LQ_TOT * DMODEL * 4);
  unsigned short* hidb = (unsigned short*)take((size_t)LQ_TOT * DFFN * 2);
  float* refp  = (float*)take((size_t)LQ_TOT * 2 * 4);
  unsigned short* wOffB = (unsigned short*)take((size_t)NLAYER * 256 * 256 * 2);
  unsigned short* wAttB = (unsigned short*)take((size_t)NLAYER * 128 * 256 * 2);
  unsigned short* wVB   = (unsigned short*)take((size_t)NLAYER * 256 * 256 * 2);
  unsigned short* wOB   = (unsigned short*)take((size_t)NLAYER * 256 * 256 * 2);
  unsigned short* wL1B  = (unsigned short*)take((size_t)NLAYER * 1024 * 256 * 2);
  unsigned short* wL2B  = (unsigned short*)take((size_t)NLAYER * 256 * 1024 * 2);
  (void)ws_size;

  // layout prep + weight bf16 conversion (recomputed every call; deterministic)
  prep_kernel<<<(LQ_TOT * DMODEL + 255) / 256, 256, 0, stream>>>(
      src0, pos0, src1, pos1, src2, pos2, src3, pos3, lev, sbuf, posem, refp);
  {
    int n;
    n = NLAYER * 256 * 256;  cvt_bf16_kernel<<<(n + 255) / 256, 256, 0, stream>>>(Woff, wOffB, n);
    n = NLAYER * 128 * 256;  cvt_bf16_kernel<<<(n + 255) / 256, 256, 0, stream>>>(Watt, wAttB, n);
    n = NLAYER * 256 * 256;  cvt_bf16_kernel<<<(n + 255) / 256, 256, 0, stream>>>(Wv, wVB, n);
    n = NLAYER * 256 * 256;  cvt_bf16_kernel<<<(n + 255) / 256, 256, 0, stream>>>(Wo, wOB, n);
    n = NLAYER * 1024 * 256; cvt_bf16_kernel<<<(n + 255) / 256, 256, 0, stream>>>(Wl1, wL1B, n);
    n = NLAYER * 256 * 1024; cvt_bf16_kernel<<<(n + 255) / 256, 256, 0, stream>>>(Wl2, wL2B, n);
  }

  const int MB = LQ_TOT / 128;  // 170 block rows
  for (int L = 0; L < NLAYER; ++L) {
    const unsigned short* woff = wOffB + (size_t)L * 256 * 256;
    const unsigned short* watt = wAttB + (size_t)L * 128 * 256;
    const unsigned short* wv   = wVB   + (size_t)L * 256 * 256;
    const unsigned short* wo   = wOB   + (size_t)L * 256 * 256;
    const unsigned short* wl1  = wL1B  + (size_t)L * 1024 * 256;
    const unsigned short* wl2  = wL2B  + (size_t)L * 256 * 1024;

    // off = (s+pos) @ Woff^T + boff     [LQ,256]
    gemm_kernel<256, 256, false, true, false, false><<<dim3(2, MB), 256, 0, stream>>>(
        sbuf, posem, woff, boff + L * 256, offb);
    // attn = (s+pos) @ Wattn^T + battn  [LQ,128]
    gemm_kernel<128, 256, false, true, false, false><<<dim3(1, MB), 256, 0, stream>>>(
        sbuf, posem, watt, batt + L * 128, attnb);
    // val = s @ Wv^T + bv               [LQ,256]
    gemm_kernel<256, 256, false, false, false, false><<<dim3(2, MB), 256, 0, stream>>>(
        sbuf, nullptr, wv, bv + L * 256, valb);

    softmax16_kernel<<<(LQ_TOT * NHEAD + 255) / 256, 256, 0, stream>>>(attnb);
    msda_kernel<<<(LQ_TOT * NHEAD * DHEAD + 255) / 256, 256, 0, stream>>>(
        valb, offb, attnb, refp, msdab);

    // a = msda @ Wo^T + bo  (into offb), then s = LN(s + a)
    gemm_kernel<256, 256, false, false, false, false><<<dim3(2, MB), 256, 0, stream>>>(
        msdab, nullptr, wo, bo + L * 256, offb);
    add_ln_kernel<<<(LQ_TOT * 32 + 255) / 256, 256, 0, stream>>>(
        sbuf, offb, g1 + L * 256, b1 + L * 256);

    // h = relu(s @ Wl1^T + bl1) as bf16; f = h @ Wl2^T + bl2; s = LN(s + f)
    gemm_kernel<1024, 256, false, false, true, true><<<dim3(8, MB), 256, 0, stream>>>(
        sbuf, nullptr, wl1, bl1 + L * 1024, hidb);
    gemm_kernel<256, 1024, true, false, false, false><<<dim3(2, MB), 256, 0, stream>>>(
        hidb, nullptr, wl2, bl2 + L * 256, offb);
    add_ln_kernel<<<(LQ_TOT * 32 + 255) / 256, 256, 0, stream>>>(
        sbuf, offb, g2 + L * 256, b2 + L * 256);
  }

  (void)hipMemcpyAsync(d_out, sbuf, (size_t)LQ_TOT * DMODEL * 4,
                       hipMemcpyDeviceToDevice, stream);
}
